// MyCLIPEncoderSqueeze_prior_87634512708109
// MI455X (gfx1250) — compile-verified, matrix operand order checked by reading, not run on hardware
//
#include <hip/hip_runtime.h>
#include <hip/hip_bf16.h>

// ---------------------------------------------------------------------------
// MI455X (gfx1250) implementation notes:
//  * wave32 only; all reductions use 32-lane shfl_xor trees.
//  * Dominant GEMMs (dist = X @ P^T  and  prior = softmax @ P, 2 x 62 GMAC)
//    run as a fused flash-softmax kernel using v_wmma_f32_16x16x32_bf16.
//    prototype (12.6 MB bf16) is L2-resident (192 MB L2); the 2x323MB
//    dist/probs intermediates never hit HBM.
//  * prototype pre-converted to bf16 in BOTH row-major and transposed layouts
//    so both GEMMs load contiguous 32B per-lane vectors.
//  * LDS tile staging uses GLOBAL_LOAD_ASYNC_TO_LDS_B128 (ASYNCcnt +
//    s_wait_asynccnt); builtin confirmed present, parameter type is
//    int __attribute__((vector_size(16))) * in AS1/AS3.
//  * Gumbel noise: deterministic counter-hash RNG (NOT bit-exact JAX
//    threefry; no runtime check available in this loop).
// ---------------------------------------------------------------------------

typedef __attribute__((ext_vector_type(16))) __bf16 v16bf;
typedef __attribute__((ext_vector_type(8)))  float  v8f;
typedef int v4i_b128 __attribute__((vector_size(16)));   // builtin's pointee

#define B_SZ   128
#define D_DIM  768
#define P_DIM  8192
#define SEQ    77
#define MROWS  (B_SZ * SEQ)      // 9856 = 616 * 16
#define CHUNK  64                // prototypes per flash step
#define CCHUNK 16                // conv input-channel LDS chunk

// ---- CDNA5 async global->LDS path (guarded; fallback = uint4 copies) ------
#if defined(__has_builtin)
#if __has_builtin(__builtin_amdgcn_global_load_async_to_lds_b128) && \
    __has_builtin(__builtin_amdgcn_s_wait_asynccnt)
#define HAVE_ASYNC_LDS 1
#endif
#endif
#ifndef HAVE_ASYNC_LDS
#define HAVE_ASYNC_LDS 0
#endif

__device__ __forceinline__ void lds_copy16(const void* g, void* l) {
#if HAVE_ASYNC_LDS
  __builtin_amdgcn_global_load_async_to_lds_b128(
      (__attribute__((address_space(1))) v4i_b128*)g,
      (__attribute__((address_space(3))) v4i_b128*)l, 0, 0);
#else
  *(uint4*)l = *(const uint4*)g;
#endif
}
__device__ __forceinline__ void lds_copy_fence() {
#if HAVE_ASYNC_LDS
  __builtin_amdgcn_s_wait_asynccnt(0);
#endif
}

__device__ __forceinline__ __bf16 f32_to_bf16(float f) {
  union { float f; unsigned u; } v; v.f = f;
  unsigned r = v.u + 0x7fffu + ((v.u >> 16) & 1u);   // RNE
  union { unsigned short s; __bf16 b; } o; o.s = (unsigned short)(r >> 16);
  return o.b;
}
__device__ __forceinline__ float bf16_to_f32(__bf16 b) {
  union { unsigned short s; __bf16 b; } i; i.b = b;
  union { unsigned u; float f; } o; o.u = ((unsigned)i.s) << 16;
  return o.f;
}
__device__ __forceinline__ float wave_sum(float s) {
  for (int off = 16; off; off >>= 1) s += __shfl_xor(s, off, 32);
  return s;
}
__device__ __forceinline__ float gumbel_noise(unsigned row, unsigned col) {
  unsigned x = row * 0x9E3779B9u ^ (col + 0x6A09E667u) * 0x85EBCA6Bu;
  x ^= x >> 16; x *= 0x7FEB352Du; x ^= x >> 15; x *= 0x846CA68Bu; x ^= x >> 16;
  float u = (float)(x >> 8) * (1.0f / 16777216.0f) + (0.5f / 16777216.0f);
  return -__logf(-__logf(u));
}

// ---------------------------------------------------------------------------
// bf16 conversion kernels (prototype in two layouts; wk/wv)
// ---------------------------------------------------------------------------
__global__ void cvt_proto_kernel(const float* __restrict__ proto,
                                 __bf16* __restrict__ pb,
                                 __bf16* __restrict__ pTb) {
  size_t i = (size_t)blockIdx.x * blockDim.x + threadIdx.x;
  if (i >= (size_t)P_DIM * D_DIM) return;
  unsigned r = (unsigned)(i / D_DIM), c = (unsigned)(i % D_DIM);
  __bf16 v = f32_to_bf16(proto[i]);
  pb[i] = v;
  pTb[(size_t)c * P_DIM + r] = v;
}

__global__ void cvt_mat_kernel(const float* __restrict__ a,
                               __bf16* __restrict__ o, int n) {
  int i = blockIdx.x * blockDim.x + threadIdx.x;
  if (i < n) o[i] = f32_to_bf16(a[i]);
}

// ---------------------------------------------------------------------------
// conv1d (K=3, SAME, NCH) + LayerNorm(axis=-1) + ReLU.
// One wave owns one full output row (768 positions => 24 per lane), so the
// LN mean/var reduce entirely inside the wave. Input channels staged through
// LDS in chunks of 16 (48 KB) shared by the 4 waves of the block, using
// async global->LDS when available.
// MODE 0: conv1 (in = concat(text_features, text_latent), Cout=257 -> out32)
// MODE 1: conv2 (in = x1, compute channels 1..77 only -> out32 + outbf)
// ---------------------------------------------------------------------------
template <int MODE>
__global__ __launch_bounds__(128) void conv_ln_relu(
    const float* __restrict__ in0, const float* __restrict__ in1,
    const float* __restrict__ w,  const float* __restrict__ cb,
    const float* __restrict__ g,  const float* __restrict__ beta,
    float* __restrict__ out32, __bf16* __restrict__ outbf) {
  constexpr int Cin = 257;
  __shared__ __attribute__((aligned(16))) float ldsIn[CCHUNK * D_DIM]; // 48 KB

  const int tid  = threadIdx.x;
  const int wave = tid >> 5;
  const int lane = tid & 31;
  const int b    = blockIdx.x;
  const int o    = (MODE == 0) ? (blockIdx.y * 4 + wave)
                               : (1 + blockIdx.y * 4 + wave);
  const bool valid = (MODE == 0) ? (o < 257) : (o <= SEQ);
  const int h0 = lane * 24;

  float acc[24];
#pragma unroll
  for (int t = 0; t < 24; t++) acc[t] = 0.f;

  for (int ic0 = 0; ic0 < Cin; ic0 += CCHUNK) {
    const int cnt = min(CCHUNK, Cin - ic0);
    // stage cnt input rows (float4 granularity, 192 per row)
    for (int i = tid; i < cnt * 192; i += 128) {
      const int ic = ic0 + i / 192;
      const int h4 = (i % 192) * 4;
      const float* src;
      if (MODE == 0) {
        src = (ic == 0) ? in0 + (size_t)b * D_DIM + h4
                        : in1 + ((size_t)b * 256 + (ic - 1)) * D_DIM + h4;
      } else {
        src = in0 + ((size_t)b * 257 + ic) * D_DIM + h4;
      }
      lds_copy16(src, ldsIn + i * 4);
    }
    lds_copy_fence();
    __syncthreads();
    if (valid) {
      for (int j = 0; j < cnt; j++) {
        const float* wr = w + ((size_t)o * Cin + ic0 + j) * 3;
        const float w0 = wr[0], w1 = wr[1], w2 = wr[2];
        const float* row = ldsIn + j * D_DIM;
        float win[26];
#pragma unroll
        for (int t = 0; t < 26; t++) {
          const int h = h0 - 1 + t;
          win[t] = (h >= 0 && h < D_DIM) ? row[h] : 0.f;
        }
#pragma unroll
        for (int t = 0; t < 24; t++)
          acc[t] += w0 * win[t] + w1 * win[t + 1] + w2 * win[t + 2];
      }
    }
    __syncthreads();
  }
  if (!valid) return;

  const float bias = cb[o];
  float s = 0.f;
#pragma unroll
  for (int t = 0; t < 24; t++) { acc[t] += bias; s += acc[t]; }
  const float mean = wave_sum(s) * (1.0f / 768.0f);
  float v2 = 0.f;
#pragma unroll
  for (int t = 0; t < 24; t++) { const float d = acc[t] - mean; v2 += d * d; }
  const float inv = rsqrtf(wave_sum(v2) * (1.0f / 768.0f) + 1e-5f);

  const size_t row_out = (MODE == 0) ? ((size_t)b * 257 + o)
                                     : ((size_t)b * SEQ + (o - 1));
#pragma unroll
  for (int t = 0; t < 24; t++) {
    const int h = h0 + t;
    float y = (acc[t] - mean) * inv * g[h] + beta[h];
    y = fmaxf(y, 0.f);
    if (out32) out32[row_out * D_DIM + h] = y;
    if (outbf) outbf[row_out * D_DIM + h] = f32_to_bf16(y);
  }
}

// ---------------------------------------------------------------------------
// Fused prototype quantization (flash-softmax), the FLOP-dominant stage.
// Block = 4 waves, one 16-row M tile of Xp (A staged once in LDS).
// Per 64-prototype chunk:
//   GEMM1 (wmma bf16): 16x16 dist tile per wave, K=768 (24 wmma)
//   + gumbel, /TAU -> online softmax state (m,l), 8 threads per row
//   GEMM2 (wmma bf16): acc(16x768) += probs(16x64) @ proto[chunk], B tiles
//   double-buffered so loads overlap the WMMA pipe.
// Final: acc / l -> image_prior (bf16).
// ---------------------------------------------------------------------------
__global__ __launch_bounds__(128) void quantize_kernel(
    const __bf16* __restrict__ Xp, const __bf16* __restrict__ proto,
    const __bf16* __restrict__ protoT, __bf16* __restrict__ ip_out) {
  __shared__ __attribute__((aligned(32))) __bf16 ldsA[16 * D_DIM];   // 24 KB
  __shared__ __attribute__((aligned(16))) float  ldsDist[16 * CHUNK];
  __shared__ __attribute__((aligned(32))) __bf16 ldsProbs[16 * CHUNK];
  __shared__ float ldsM[16], ldsL[16], ldsC[16];

  const int tid  = threadIdx.x;
  const int wave = tid >> 5;
  const int lane = tid & 31;
  const int m0   = blockIdx.x * 16;
  const int colA = lane & 15;       // A-row (M) / C-col (N) per WMMA layout
  const int half = lane >> 4;       // K half-select for 16-bit A/B layout

  // Stage A tile: 16 rows x 768 bf16 (24 KB), async global->LDS
  {
    const uint4* src = (const uint4*)(Xp + (size_t)m0 * D_DIM);
    uint4* dst = (uint4*)ldsA;
    for (int i = tid; i < (16 * D_DIM * 2) / 16; i += 128)
      lds_copy16(src + i, dst + i);
    lds_copy_fence();
  }
  if (tid < 16) { ldsM[tid] = -1e30f; ldsL[tid] = 0.f; }
  __syncthreads();

  // Wave owns output columns [wave*192, wave*192+192): 12 f32 16x16 tiles
  v8f acc[12];
  const v8f vzero = {0.f, 0.f, 0.f, 0.f, 0.f, 0.f, 0.f, 0.f};
#pragma unroll
  for (int t = 0; t < 12; t++) acc[t] = vzero;

  for (int p0 = 0; p0 < P_DIM; p0 += CHUNK) {
    // -------- GEMM1: dist 16x16 at prototype cols p0 + wave*16 --------
    v8f dt = vzero;
    const __bf16* Brow = proto + (size_t)(p0 + wave * 16 + colA) * D_DIM;
    __builtin_prefetch(Brow + D_DIM * CHUNK, 0, 1);   // next chunk (L2)
#pragma unroll 4
    for (int k0 = 0; k0 < D_DIM; k0 += 32) {
      v16bf a = *(const v16bf*)(ldsA + colA * D_DIM + k0 + half * 16);
      v16bf b = *(const v16bf*)(Brow + k0 + half * 16);
      dt = __builtin_amdgcn_wmma_f32_16x16x32_bf16(false, a, false, b,
                                                   (short)0, dt, false, false);
    }
    // dist + gumbel, scaled by 1/TAU -> LDS (C layout: row=i+(lane<16?0:8))
    {
      const int rb = (lane < 16) ? 0 : 8;
#pragma unroll
      for (int i = 0; i < 8; i++) {
        const int r = rb + i;
        const float v = dt[i] + gumbel_noise(m0 + r, p0 + wave * 16 + colA);
        ldsDist[r * CHUNK + wave * 16 + colA] = v * (1.0f / 0.9f);
      }
    }
    __syncthreads();
    // -------- online softmax update: 8 threads per row (all 128 active) ---
    {
      const int r   = tid >> 3;         // 0..15
      const int sub = tid & 7;          // 8-lane group inside a wave
      const float mOld = ldsM[r];
      float vals[8];
      float lmax = -1e30f;
#pragma unroll
      for (int c = 0; c < 8; c++) {
        vals[c] = ldsDist[r * CHUNK + sub * 8 + c];
        lmax = fmaxf(lmax, vals[c]);
      }
#pragma unroll
      for (int msk = 1; msk <= 4; msk <<= 1)
        lmax = fmaxf(lmax, __shfl_xor(lmax, msk, 32));
      const float mNew = fmaxf(mOld, lmax);
      float s = 0.f;
#pragma unroll
      for (int c = 0; c < 8; c++) {
        const float e = __expf(vals[c] - mNew);
        s += e;
        ldsProbs[r * CHUNK + sub * 8 + c] = f32_to_bf16(e);
      }
#pragma unroll
      for (int msk = 1; msk <= 4; msk <<= 1) s += __shfl_xor(s, msk, 32);
      if (sub == 0) {
        const float corr = __expf(mOld - mNew);
        ldsC[r] = corr;
        ldsM[r] = mNew;
        ldsL[r] = ldsL[r] * corr + s;
      }
    }
    __syncthreads();
    // -------- rescale running accumulators --------
    {
      const int rb = (lane < 16) ? 0 : 8;
      float cf[8];
#pragma unroll
      for (int i = 0; i < 8; i++) cf[i] = ldsC[rb + i];
#pragma unroll
      for (int t = 0; t < 12; t++)
#pragma unroll
        for (int i = 0; i < 8; i++) acc[t][i] *= cf[i];
    }
    // -------- GEMM2: acc += probs(16x64) @ proto[p0:p0+64, :] --------
    // B from transposed prototype: contiguous per-lane k. Double-buffered
    // so B loads for tile t+1 overlap the WMMA issue of tile t.
    {
      v16bf a0 = *(const v16bf*)(ldsProbs + colA * CHUNK + half * 16);
      v16bf a1 = *(const v16bf*)(ldsProbs + colA * CHUNK + 32 + half * 16);
      const __bf16* Bt = protoT + (size_t)(wave * 192 + colA) * P_DIM + p0 +
                         half * 16;
      v16bf bn0 = *(const v16bf*)(Bt);
      v16bf bn1 = *(const v16bf*)(Bt + 32);
#pragma unroll
      for (int t = 0; t < 12; t++) {
        v16bf c0 = bn0, c1 = bn1;
        if (t < 11) {
          const __bf16* Bn = Bt + (size_t)(t + 1) * 16 * P_DIM;
          bn0 = *(const v16bf*)(Bn);
          bn1 = *(const v16bf*)(Bn + 32);
        }
        acc[t] = __builtin_amdgcn_wmma_f32_16x16x32_bf16(
            false, a0, false, c0, (short)0, acc[t], false, false);
        acc[t] = __builtin_amdgcn_wmma_f32_16x16x32_bf16(
            false, a1, false, c1, (short)0, acc[t], false, false);
      }
    }
    __syncthreads();  // protect ldsDist/ldsProbs for next chunk
  }

  // -------- finalize: divide by softmax denominator, store bf16 --------
  {
    const int rb = (lane < 16) ? 0 : 8;
    float invl[8];
#pragma unroll
    for (int i = 0; i < 8; i++) invl[i] = 1.0f / ldsL[rb + i];
#pragma unroll
    for (int t = 0; t < 12; t++) {
      const int n0 = wave * 192 + t * 16;
#pragma unroll
      for (int i = 0; i < 8; i++) {
        ip_out[(size_t)(m0 + rb + i) * D_DIM + n0 + colA] =
            f32_to_bf16(acc[t][i] * invl[i]);
      }
    }
  }
}

// ---------------------------------------------------------------------------
// k/v projections (out = X @ W^T + b), WMMA bf16, shared LDS A tile.
// Grid: (616 M-tiles, 12 N-groups), wave w of a block handles one 16-col tile.
// ---------------------------------------------------------------------------
__global__ __launch_bounds__(128) void proj_kv_kernel(
    const __bf16* __restrict__ ip, const __bf16* __restrict__ wkb,
    const __bf16* __restrict__ wvb, const float* __restrict__ bk,
    const float* __restrict__ bv, __bf16* __restrict__ kout,
    __bf16* __restrict__ vout) {
  __shared__ __attribute__((aligned(32))) __bf16 ldsA[16 * D_DIM];  // 24 KB
  const int tid  = threadIdx.x;
  const int wave = tid >> 5;
  const int lane = tid & 31;
  const int m0   = blockIdx.x * 16;
  const int n0   = (blockIdx.y * 4 + wave) * 16;
  const int colA = lane & 15;
  const int half = lane >> 4;

  {
    const uint4* src = (const uint4*)(ip + (size_t)m0 * D_DIM);
    uint4* dst = (uint4*)ldsA;
    for (int i = tid; i < (16 * D_DIM * 2) / 16; i += 128)
      lds_copy16(src + i, dst + i);
    lds_copy_fence();
  }
  __syncthreads();

  const v8f vzero = {0.f, 0.f, 0.f, 0.f, 0.f, 0.f, 0.f, 0.f};
  v8f ka = vzero, va = vzero;
  const __bf16* Bk = wkb + (size_t)(n0 + colA) * D_DIM;
  const __bf16* Bv = wvb + (size_t)(n0 + colA) * D_DIM;
#pragma unroll 4
  for (int k0 = 0; k0 < D_DIM; k0 += 32) {
    v16bf a  = *(const v16bf*)(ldsA + colA * D_DIM + k0 + half * 16);
    v16bf b1 = *(const v16bf*)(Bk + k0 + half * 16);
    v16bf b2 = *(const v16bf*)(Bv + k0 + half * 16);
    ka = __builtin_amdgcn_wmma_f32_16x16x32_bf16(false, a, false, b1,
                                                 (short)0, ka, false, false);
    va = __builtin_amdgcn_wmma_f32_16x16x32_bf16(false, a, false, b2,
                                                 (short)0, va, false, false);
  }
  const int rb = (lane < 16) ? 0 : 8;
  const float bkc = bk[n0 + colA], bvc = bv[n0 + colA];
#pragma unroll
  for (int i = 0; i < 8; i++) {
    const size_t idx = (size_t)(m0 + rb + i) * D_DIM + n0 + colA;
    kout[idx] = f32_to_bf16(ka[i] + bkc);
    vout[idx] = f32_to_bf16(va[i] + bvc);
  }
}

// ---------------------------------------------------------------------------
// Attention + pool, only query row 0 per batch survives to the output, so
// this is a handful of matvecs per batch — pure VALU, negligible time.
// ---------------------------------------------------------------------------
__global__ __launch_bounds__(256) void attn_pool_kernel(
    const __bf16* __restrict__ ip, const __bf16* __restrict__ kb,
    const __bf16* __restrict__ vb, const float* __restrict__ wq,
    const float* __restrict__ bq, const float* __restrict__ wo,
    const float* __restrict__ bo, float* __restrict__ out) {
  __shared__ float q0[D_DIM];
  __shared__ float att[80];
  __shared__ float pooled[D_DIM];
  const int tid  = threadIdx.x;
  const int wave = tid >> 5;
  const int lane = tid & 31;
  const int b    = blockIdx.x;
  const size_t m0 = (size_t)b * SEQ;

  for (int n = tid; n < D_DIM; n += 256) {
    float s = bq[n];
    const __bf16* x = ip + m0 * D_DIM;
    const float* wr = wq + (size_t)n * D_DIM;
    for (int k = 0; k < D_DIM; k++) s += bf16_to_f32(x[k]) * wr[k];
    q0[n] = s;
  }
  __syncthreads();

  const float scale = 0.03608439182435161f;  // 1/sqrt(768)
  for (int j = wave; j < SEQ; j += 8) {
    const __bf16* kr = kb + (m0 + j) * D_DIM;
    float s = 0.f;
    for (int k = lane; k < D_DIM; k += 32) s += q0[k] * bf16_to_f32(kr[k]);
    s = wave_sum(s);
    if (lane == 0) att[j] = s * scale;
  }
  __syncthreads();

  if (wave == 0) {
    float mx = -1e30f;
    for (int j = lane; j < SEQ; j += 32) mx = fmaxf(mx, att[j]);
    for (int off = 16; off; off >>= 1) mx = fmaxf(mx, __shfl_xor(mx, off, 32));
    float e[3]; int cnt = 0; float s = 0.f;
    for (int j = lane; j < SEQ; j += 32) { e[cnt] = __expf(att[j] - mx); s += e[cnt]; cnt++; }
    s = wave_sum(s);
    const float invs = 1.0f / s;
    cnt = 0;
    for (int j = lane; j < SEQ; j += 32) att[j] = e[cnt++] * invs;
  }
  __syncthreads();

  for (int n = tid; n < D_DIM; n += 256) {
    float s = 0.f;
    for (int j = 0; j < SEQ; j++) s += att[j] * bf16_to_f32(vb[(m0 + j) * D_DIM + n]);
    pooled[n] = s;
  }
  __syncthreads();

  for (int n = tid; n < D_DIM; n += 256) {
    float s = bo[n];
    const float* wr = wo + (size_t)n * D_DIM;
    for (int k = 0; k < D_DIM; k++) s += pooled[k] * wr[k];
    out[(size_t)b * D_DIM + n] = s;
  }
}

// ---------------------------------------------------------------------------
// Host launcher. Workspace layout (bytes, all 256-aligned):
//   [0,        101056512)  x1 f32 (128x257x768)   -- dead after conv2, reused:
//       [0,        15138816)  image_prior bf16 (9856x768)
//       [15138816, 30277632)  k bf16
//       [30277632, 45416448)  v bf16
//   [101056512, 116195328)  Xp bf16 (9856x768)
//   [116195328, 128778240)  prototype bf16 row-major
//   [128778240, 141361152)  prototype bf16 transposed (768x8192)
//   [141361152, 142540800)  wk bf16
//   [142540800, 143720448)  wv bf16
// ---------------------------------------------------------------------------
extern "C" void kernel_launch(void* const* d_in, const int* in_sizes, int n_in,
                              void* d_out, int out_size, void* d_ws,
                              size_t ws_size, hipStream_t stream) {
  const float* text_features = (const float*)d_in[0];
  const float* text_latent   = (const float*)d_in[1];
  const float* conv1_w = (const float*)d_in[2];
  const float* conv1_b = (const float*)d_in[3];
  const float* ln1_g   = (const float*)d_in[4];
  const float* ln1_b   = (const float*)d_in[5];
  const float* conv2_w = (const float*)d_in[6];
  const float* conv2_b = (const float*)d_in[7];
  const float* ln2_g   = (const float*)d_in[8];
  const float* ln2_b   = (const float*)d_in[9];
  const float* prototype = (const float*)d_in[10];
  const float* wq = (const float*)d_in[11];
  const float* bq = (const float*)d_in[12];
  const float* wk = (const float*)d_in[13];
  const float* bk = (const float*)d_in[14];
  const float* wv = (const float*)d_in[15];
  const float* bv = (const float*)d_in[16];
  const float* wo = (const float*)d_in[17];
  const float* bo = (const float*)d_in[18];

  char* ws = (char*)d_ws;
  float*  x1    = (float*)(ws);
  __bf16* ip_bf = (__bf16*)(ws);                 // reuse of x1 region
  __bf16* kb    = (__bf16*)(ws + 15138816);
  __bf16* vb    = (__bf16*)(ws + 30277632);
  __bf16* Xp    = (__bf16*)(ws + 101056512);
  __bf16* pb    = (__bf16*)(ws + 116195328);
  __bf16* pTb   = (__bf16*)(ws + 128778240);
  __bf16* wkb   = (__bf16*)(ws + 141361152);
  __bf16* wvb   = (__bf16*)(ws + 142540800);

  float* patches_out = (float*)d_out;                       // 128*77*768
  float* prior_out   = (float*)d_out + (size_t)MROWS * D_DIM;

  // bf16 conversions (independent of convs, overlap on the queue)
  cvt_proto_kernel<<<(P_DIM * D_DIM + 255) / 256, 256, 0, stream>>>(prototype, pb, pTb);
  cvt_mat_kernel<<<(D_DIM * D_DIM + 255) / 256, 256, 0, stream>>>(wk, wkb, D_DIM * D_DIM);
  cvt_mat_kernel<<<(D_DIM * D_DIM + 255) / 256, 256, 0, stream>>>(wv, wvb, D_DIM * D_DIM);

  // conv1 + LN + ReLU  -> x1 (all 257 channels)
  conv_ln_relu<0><<<dim3(B_SZ, 65), 128, 0, stream>>>(
      text_features, text_latent, conv1_w, conv1_b, ln1_g, ln1_b, x1, nullptr);

  // conv2 + LN + ReLU -> text_patches (f32 output) + Xp (bf16); LN is per-row
  // so channel 0 (sliced away by the reference) is never computed.
  conv_ln_relu<1><<<dim3(B_SZ, 20), 128, 0, stream>>>(
      x1, nullptr, conv2_w, conv2_b, ln2_g, ln2_b, patches_out, Xp);

  // fused dist -> gumbel-softmax -> @prototype (flash style, WMMA bf16)
  quantize_kernel<<<MROWS / 16, 128, 0, stream>>>(Xp, pb, pTb, ip_bf);

  // k/v projections (WMMA bf16)
  proj_kv_kernel<<<dim3(MROWS / 16, 12), 128, 0, stream>>>(
      ip_bf, wkb, wvb, bk, bv, kb, vb);

  // attention restricted to query row 0 per batch + output projection
  attn_pool_kernel<<<B_SZ, 256, 0, stream>>>(ip_bf, kb, vb, wq, bq, wo, bo,
                                             prior_out);
}
